// ExportEquiDeformConv_31044023615887
// MI455X (gfx1250) — matrix operand order repaired and checked by reading, not compiled, not run on hardware
//
#include <hip/hip_runtime.h>
#include <hip/hip_bf16.h>
#include <math.h>

typedef __attribute__((ext_vector_type(16))) _Float16 v16h;
typedef __attribute__((ext_vector_type(8)))  float    v8f;

#define B_  4
#define C_  64
#define H_  128
#define W_  128
#define CO_ 64
// reduction dim = C_*9 = 576 = 18 chunks of K=32

// ---------------------------------------------------------------------------
// Kernel 0: NCHW -> NHWC transpose so channel gathers are contiguous.
// ---------------------------------------------------------------------------
__global__ void k_transpose(const float* __restrict__ x, float* __restrict__ xT) {
    int idx = blockIdx.x * 256 + threadIdx.x;   // < B*C*H*W = 4*2^20
    int b = idx >> 20;                          // C*H*W = 2^20
    int r = idx & ((1 << 20) - 1);
    int c = r >> 14;                            // H*W = 2^14
    int p = r & 16383;
    xT[(((size_t)b << 14) + p) * 64 + c] = x[idx];
}

// ---------------------------------------------------------------------------
// Kernel 1: 3x3 conv producing the 4-channel offset field. w_off in LDS.
// ---------------------------------------------------------------------------
__global__ void k_offconv(const float* __restrict__ xT, const float* __restrict__ w_off,
                          const float* __restrict__ b_off, float* __restrict__ off) {
    __shared__ float sw[4 * 64 * 9];
    for (int t = threadIdx.x; t < 4 * 64 * 9; t += 256) sw[t] = w_off[t];
    __syncthreads();

    int idx = blockIdx.x * 256 + threadIdx.x;   // < B*H*W = 65536
    int b = idx >> 14;
    int p = idx & 16383;
    int i = p >> 7, j = p & 127;

    float a0 = b_off[0], a1 = b_off[1], a2 = b_off[2], a3 = b_off[3];
    #pragma unroll
    for (int dy = 0; dy < 3; ++dy) {
        int yy = i + dy - 1;
        if (yy < 0 || yy >= H_) continue;
        #pragma unroll
        for (int dx = 0; dx < 3; ++dx) {
            int xx = j + dx - 1;
            if (xx < 0 || xx >= W_) continue;
            const float* px = xT + ((((size_t)b << 7) + yy) * W_ + xx) * 64;
            int tap = dy * 3 + dx;
            for (int c = 0; c < 64; ++c) {
                float xv = px[c];
                a0 += xv * sw[(0 * 64 + c) * 9 + tap];
                a1 += xv * sw[(1 * 64 + c) * 9 + tap];
                a2 += xv * sw[(2 * 64 + c) * 9 + tap];
                a3 += xv * sw[(3 * 64 + c) * 9 + tap];
            }
        }
    }
    size_t base = (((size_t)b * 4) << 14) + p;   // off laid out [B,4,H,W]
    off[base]                 = a0;
    off[base + 1 * 16384]     = a1;
    off[base + 2 * 16384]     = a2;
    off[base + 3 * 16384]     = a3;
}

// ---------------------------------------------------------------------------
// Kernel 2: pack weight [Co=64, C=64, 3, 3] (f32) into f16 WMMA B-fragments.
// Reduction order: idx = tap*64 + c, split into 18 chunks kk of K=32
// (tap = kk>>1, c = 32*(kk&1) + kq). B 32x16 f16 layout (wave32):
//   lane l: N = l&15 ; element e (half within v16h): kq = 16*(e>>3) + 8*(l>>4) + (e&7).
// Flat: wf16[((kk*4+nt)*32 + l)*16 + e]
// ---------------------------------------------------------------------------
__global__ void k_wfrag(const float* __restrict__ weight, _Float16* __restrict__ wf16) {
    int f  = blockIdx.x * 256 + threadIdx.x;    // < 18*4*32*16 = 36864
    int e  = f & 15;
    int l  = (f >> 4) & 31;
    int nt = (f >> 9) & 3;
    int kk = f >> 11;
    int n   = nt * 16 + (l & 15);
    int kq  = ((e >> 3) << 4) + ((l >> 4) << 3) + (e & 7);
    int tap = kk >> 1;
    int c   = ((kk & 1) << 5) + kq;
    wf16[f] = (_Float16)weight[(n * 64 + c) * 9 + tap];
}

// ---------------------------------------------------------------------------
// Kernel 3: deformable-conv implicit GEMM with WMMA f32_16x16x32_f16.
// Block = one (b, row i): 8 waves x 16 pixels = 128 pixels; each wave
// computes a 16(M=pixels) x 64(N=out channels) tile as 4 accumulators.
// A fragment (16x32 f16): lane m = lane&15 is the pixel; group g = lane>>4
// selects K halves per ISA layout; per-lane channels are contiguous in NHWC
// so corners are read as float4s and blended with bilinear weights in f32.
// ---------------------------------------------------------------------------
__global__ void __launch_bounds__(256)
k_deform_wmma(const float* __restrict__ xT, const float* __restrict__ off,
              const _Float16* __restrict__ wf16, const float* __restrict__ bias,
              float* __restrict__ out) {
    const int tid  = threadIdx.x;
    const int wave = tid >> 5;
    const int lane = tid & 31;
    const int g    = lane >> 4;
    const int m    = lane & 15;
    const int blk  = blockIdx.x;        // b*H + i
    const int b    = blk >> 7;
    const int i    = blk & 127;
    const int j    = wave * 16 + m;     // this lane's pixel column

    // Per-pixel offset parameters (off layout [B,4,H,W]).
    const size_t pbase = (((size_t)b * 4) << 14) + ((size_t)i << 7) + j;
    const float o0 = off[pbase];
    const float o1 = off[pbase + 16384];
    float s1 = off[pbase + 32768]; s1 = (s1 > 0.f ? s1 : 0.f) + 1.f;
    float s2 = off[pbase + 49152]; s2 = (s2 > 0.f ? s2 : 0.f) + 1.f;

    v8f acc[4];
    #pragma unroll
    for (int nt = 0; nt < 4; ++nt) {
        const float bn = bias[nt * 16 + m];   // N = nt*16 + (lane&15)
        #pragma unroll
        for (int r = 0; r < 8; ++r) acc[nt][r] = bn;
    }

    #pragma unroll
    for (int tap = 0; tap < 9; ++tap) {
        const float ky = (float)(tap / 3 - 1);
        const float kx = (float)(tap % 3 - 1);
        const float ys = (float)i - 1.f + ky * s1 + o0;
        const float xs = (float)j - 1.f + kx * s2 + o1;
        const float y0f = floorf(ys), x0f = floorf(xs);
        const float wy = ys - y0f, wx = xs - x0f;
        const int y0 = (int)y0f, x0 = (int)x0f;
        const int y1 = y0 + 1,  x1 = x0 + 1;
        const bool vy0 = (y0 >= 0) && (y0 < H_);
        const bool vy1 = (y1 >= 0) && (y1 < H_);
        const bool vx0 = (x0 >= 0) && (x0 < W_);
        const bool vx1 = (x1 >= 0) && (x1 < W_);
        const float f00 = (vy0 && vx0) ? (1.f - wy) * (1.f - wx) : 0.f;
        const float f01 = (vy0 && vx1) ? (1.f - wy) * wx         : 0.f;
        const float f10 = (vy1 && vx0) ? wy * (1.f - wx)         : 0.f;
        const float f11 = (vy1 && vx1) ? wy * wx                 : 0.f;
        const int yc0 = vy0 ? y0 : 0, yc1 = vy1 ? y1 : 0;
        const int xc0 = vx0 ? x0 : 0, xc1 = vx1 ? x1 : 0;
        const float* p00 = xT + (((((size_t)b << 7) + yc0) << 7) + xc0) * 64;
        const float* p01 = xT + (((((size_t)b << 7) + yc0) << 7) + xc1) * 64;
        const float* p10 = xT + (((((size_t)b << 7) + yc1) << 7) + xc0) * 64;
        const float* p11 = xT + (((((size_t)b << 7) + yc1) << 7) + xc1) * 64;

        #pragma unroll
        for (int hh = 0; hh < 2; ++hh) {      // channel half: c in [32*hh, 32*hh+32)
            v16h a = {};
            #pragma unroll
            for (int seg = 0; seg < 2; ++seg) {
                // A-layout: lane group g, element e -> channel 8*(2*seg+g) + (e&7)
                const int c0 = hh * 32 + 8 * (2 * seg + g);
                const float4 u00a = *(const float4*)(p00 + c0);
                const float4 u00b = *(const float4*)(p00 + c0 + 4);
                const float4 u01a = *(const float4*)(p01 + c0);
                const float4 u01b = *(const float4*)(p01 + c0 + 4);
                const float4 u10a = *(const float4*)(p10 + c0);
                const float4 u10b = *(const float4*)(p10 + c0 + 4);
                const float4 u11a = *(const float4*)(p11 + c0);
                const float4 u11b = *(const float4*)(p11 + c0 + 4);
                a[8*seg+0] = (_Float16)(f00*u00a.x + f01*u01a.x + f10*u10a.x + f11*u11a.x);
                a[8*seg+1] = (_Float16)(f00*u00a.y + f01*u01a.y + f10*u10a.y + f11*u11a.y);
                a[8*seg+2] = (_Float16)(f00*u00a.z + f01*u01a.z + f10*u10a.z + f11*u11a.z);
                a[8*seg+3] = (_Float16)(f00*u00a.w + f01*u01a.w + f10*u10a.w + f11*u11a.w);
                a[8*seg+4] = (_Float16)(f00*u00b.x + f01*u01b.x + f10*u10b.x + f11*u11b.x);
                a[8*seg+5] = (_Float16)(f00*u00b.y + f01*u01b.y + f10*u10b.y + f11*u11b.y);
                a[8*seg+6] = (_Float16)(f00*u00b.z + f01*u01b.z + f10*u10b.z + f11*u11b.z);
                a[8*seg+7] = (_Float16)(f00*u00b.w + f01*u01b.w + f10*u10b.w + f11*u11b.w);
            }
            const int kk = tap * 2 + hh;
            #pragma unroll
            for (int nt = 0; nt < 4; ++nt) {
                const v16h bfrag =
                    *(const v16h*)(wf16 + ((size_t)((kk * 4 + nt) * 32 + lane) << 4));
                acc[nt] = __builtin_amdgcn_wmma_f32_16x16x32_f16(
                    false, a, false, bfrag, (short)0, acc[nt], false, false);
            }
        }
    }

    // C/D layout: N = lane&15 (per ntile), M = r + 8*g -> output column j0+M.
    #pragma unroll
    for (int nt = 0; nt < 4; ++nt) {
        const int n = nt * 16 + m;
        float* po = out + ((((size_t)b * CO_ + n) << 7) + i) * (size_t)W_
                        + wave * 16 + 8 * g;
        #pragma unroll
        for (int r = 0; r < 8; ++r) po[r] = acc[nt][r];
    }
}

// ---------------------------------------------------------------------------
extern "C" void kernel_launch(void* const* d_in, const int* in_sizes, int n_in,
                              void* d_out, int out_size, void* d_ws, size_t ws_size,
                              hipStream_t stream) {
    const float* x      = (const float*)d_in[0];   // [4,64,128,128]
    const float* w_off  = (const float*)d_in[1];   // [4,64,3,3]
    const float* b_off  = (const float*)d_in[2];   // [4]
    const float* weight = (const float*)d_in[3];   // [64,64,3,3]
    const float* bias   = (const float*)d_in[4];   // [64]
    // d_in[5..7] (base_offset, mask1, mask2) are folded analytically.
    float* out = (float*)d_out;

    float*    xT   = (float*)d_ws;                            // 16 MB NHWC image
    float*    off  = xT + (size_t)B_ * H_ * W_ * C_;          // 1 MB offset field
    _Float16* wf16 = (_Float16*)(off + (size_t)B_ * 4 * H_ * W_); // 72 KB B-fragments

    k_transpose  <<<(B_ * C_ * H_ * W_) / 256, 256, 0, stream>>>(x, xT);
    k_offconv    <<<(B_ * H_ * W_) / 256,      256, 0, stream>>>(xT, w_off, b_off, off);
    k_wfrag      <<<(18 * 4 * 32 * 16) / 256,  256, 0, stream>>>(weight, wf16);
    k_deform_wmma<<<B_ * H_,                   256, 0, stream>>>(xT, off, wf16, bias, out);
}
